// MDLSTM_RNN_41695542509785
// MI455X (gfx1250) — matrix operand order repaired and checked by reading, not compiled, not run on hardware
//
#include <hip/hip_runtime.h>
#include <hip/hip_bf16.h>
#include <math.h>

typedef __attribute__((ext_vector_type(16))) _Float16 v16h;
typedef __attribute__((ext_vector_type(8)))  _Float16 v8h;
typedef __attribute__((ext_vector_type(8)))  float    v8f;

#define BDIM 256

__device__ __forceinline__ float sigf(float x) { return 1.0f / (1.0f + __expf(-x)); }

// ---------------------------------------------------------------------------
// Pack [Wx ; Whu ; Whl] (fp32) -> fp16 in WMMA B-fragment order:
//   Wp[dir][kc][nt][lane][e], e=0..15
//   lane<16 : col = nt*16+lane,    k = kc*32 + e        (K 0..15)
//   lane>=16: col = nt*16+lane-16, k = kc*32 + 16 + e   (K 16..31)
// so each lane's 16 halfs are 32 contiguous bytes -> 2x global_load_b128.
// ---------------------------------------------------------------------------
__global__ void pack_weights_frag(const float* __restrict__ Wx, const float* __restrict__ Whu,
                                  const float* __restrict__ Whl, _Float16* __restrict__ Wp,
                                  int Cin, int G, int Ktot, int KC, int Ntiles) {
    int N5 = 5 * G;
    int total = 4 * KC * Ntiles * 32 * 16;
    for (int idx = blockIdx.x * blockDim.x + threadIdx.x; idx < total;
         idx += gridDim.x * blockDim.x) {
        int e    = idx & 15;
        int lane = (idx >> 4) & 31;
        int nt   = (idx >> 9) % Ntiles;
        int kc   = (idx / (512 * Ntiles)) % KC;
        int d    =  idx / (512 * Ntiles * KC);
        int col  = nt * 16 + (lane & 15);
        int k    = kc * 32 + (lane >> 4) * 16 + e;
        float v = 0.0f;
        if (k < Cin)          v = Wx [(d * Cin + k) * N5 + col];
        else if (k < Cin + G) v = Whu[(d * G + (k - Cin)) * N5 + col];
        else if (k < Ktot)    v = Whl[(d * G + (k - Cin - G)) * N5 + col];
        Wp[idx] = (_Float16)v;
    }
}

// ---------------------------------------------------------------------------
// Direct conv (VALID) + optional 2x2 maxpool + tanh.
// in: NCHW fp32 [32,Ci,Hi,Wi]; out: HWBC fp16 [Ho][Wo][32][Co]
// ---------------------------------------------------------------------------
__global__ void conv_tanh_hwbc(const float* __restrict__ in, const float* __restrict__ w,
                               const float* __restrict__ bias, _Float16* __restrict__ act,
                               int Ci, int Co, int kh, int kw, int Hi, int Wi,
                               int Ho, int Wo, int pool) {
    int total = Ho * Wo * 32 * Co;
    int idx = blockIdx.x * blockDim.x + threadIdx.x;
    if (idx >= total) return;
    int co = idx % Co;
    int b  = (idx / Co) % 32;
    int wo = (idx / (Co * 32)) % Wo;
    int ho =  idx / (Co * 32 * Wo);
    int P  = pool ? 2 : 1;
    float m = -3.4e38f;
    for (int py = 0; py < P; ++py) {
        for (int px = 0; px < P; ++px) {
            int hc = ho * P + py, wc = wo * P + px;
            float acc = bias[co];
            for (int ci = 0; ci < Ci; ++ci)
                for (int ky = 0; ky < kh; ++ky)
                    for (int kx = 0; kx < kw; ++kx)
                        acc += in[(((size_t)b * Ci + ci) * Hi + hc + ky) * Wi + wc + kx]
                             * w [(((size_t)co * Ci + ci) * kh + ky) * kw + kx];
            m = fmaxf(m, acc);
        }
    }
    act[idx] = (_Float16)tanhf(m);
}

// ---------------------------------------------------------------------------
// One anti-diagonal step of 4-direction MDLSTM (templated per layer config).
// Grid: (cellsOnDiag, 4 directions). Block: 256 threads = 8 waves.
// pre[B=32,5G] = [x | h_up | h_left] @ Wp + bias   via v_wmma_f32_16x16x32_f16
// A staged in LDS [32][Kpad+8] (skewed pitch, bank-conflict-free b128 reads);
// B fragments are 32B contiguous per lane (pre-shuffled).  Fused LSTM gates.
// ---------------------------------------------------------------------------
template<int Cin, int G, int Ktot, int Kpad>
__global__ void __launch_bounds__(BDIM)
mdlstm_diag(const _Float16* __restrict__ act, const _Float16* __restrict__ Wp,
            const float* __restrict__ bias, _Float16* hbuf, float* cbuf,
            int H, int W, int dd, int ilo) {
    constexpr int N5    = 5 * G;
    constexpr int KC    = Kpad / 32;
    constexpr int NT    = N5 / 16;
    constexpr int PITCH = Kpad + 8;              // 16B skew -> conflict-free ds_load_b128
    __shared__ __align__(16) _Float16 sA[32 * PITCH];
    __shared__ float sPre[16 * N5];

    const int dir = blockIdx.y;
    const int i   = ilo + blockIdx.x;
    const int j   = dd - i;
    const int fH  = (dir >> 1) & 1, fW = dir & 1;

    _Float16*       hD    = hbuf + (size_t)dir * H * W * 32 * G;
    float*          cD    = cbuf + (size_t)dir * H * W * 32 * G;
    const _Float16* WpD   = Wp   + (size_t)dir * KC * NT * 512;
    const float*    biasD = bias + dir * N5;

    const int tid  = threadIdx.x;
    const int wave = tid >> 5, lane = tid & 31;
    const int l15  = lane & 15, hi = lane >> 4;

    // ---- stage A = [ x(i,j) | h(i-1,j) | h(i,j-1) ] as [32][PITCH] fp16 in LDS
    {
        const int b  = tid >> 3;    // 0..31
        const int t8 = tid & 7;
        const int gi = fH ? (H - 1 - i) : i;
        const int gj = fW ? (W - 1 - j) : j;
        const _Float16* actRow = act + (((size_t)gi * W + gj) * 32 + b) * Cin;
        const _Float16* hupRow = (i > 0) ? hD + (((size_t)(i - 1) * W + j) * 32 + b) * G : 0;
        const _Float16* hlRow  = (j > 0) ? hD + (((size_t)i * W + (j - 1)) * 32 + b) * G : 0;
        _Float16* dst = sA + b * PITCH;
        #pragma unroll
        for (int kb = 0; kb < Kpad; kb += 8) {
            const int k = kb + t8;
            _Float16 v = (_Float16)0.0f;
            if (k < Cin)          v = actRow[k];
            else if (k < Cin + G) { if (hupRow) v = hupRow[k - Cin]; }
            else if (k < Ktot)    { if (hlRow)  v = hlRow[k - Cin - G]; }
            dst[k] = v;
        }
    }
    __syncthreads();

    #pragma unroll
    for (int mt = 0; mt < 2; ++mt) {              // two 16-row batch tiles
        const _Float16* aRow = sA + (mt * 16 + l15) * PITCH + hi * 16;
        #pragma unroll
        for (int nt = wave; nt < NT; nt += 8) {
            const int col = nt * 16 + l15;
            const float bv = biasD[col];
            v8f c = {bv, bv, bv, bv, bv, bv, bv, bv};
            #pragma unroll
            for (int kc = 0; kc < KC; ++kc) {
                const int kk = kc * 32;
                // A fragment: two contiguous 8-half runs in LDS -> 2x ds_load_b128
                v8h aLo = *(const v8h*)(aRow + kk);
                v8h aHi = *(const v8h*)(aRow + kk + 8);
                v16h a = __builtin_shufflevector(aLo, aHi,
                         0,1,2,3,4,5,6,7,8,9,10,11,12,13,14,15);
                // B fragment: 32 contiguous bytes per lane -> 2x global_load_b128
                const v8h* bp = (const v8h*)(WpD + (((size_t)kc * NT + nt) * 32 + lane) * 16);
                v8h bLo = bp[0];
                v8h bHi = bp[1];
                v16h bm = __builtin_shufflevector(bLo, bHi,
                          0,1,2,3,4,5,6,7,8,9,10,11,12,13,14,15);
                if (kc + 1 < KC)
                    __builtin_prefetch(WpD + (((size_t)(kc + 1) * NT + nt) * 32 + lane) * 16, 0, 1);
                c = __builtin_amdgcn_wmma_f32_16x16x32_f16(
                        false, a, false, bm, (short)0, c, false, false);
            }
            // D: vgpr r -> row (hi?8:0)+r, col
            const int rowLoc = hi * 8;
            #pragma unroll
            for (int r = 0; r < 8; ++r)
                sPre[(rowLoc + r) * N5 + col] = c[r];
        }
        __syncthreads();

        // ---- LSTM gates for batch rows mt*16 .. mt*16+15 (coalesced over g)
        #pragma unroll
        for (int br = wave; br < 16; br += 8) {
            const int b = mt * 16 + br;
            const float* pr = sPre + br * N5;
            const float* cuRow = (i > 0) ? cD + (((size_t)(i - 1) * W + j) * 32 + b) * G : 0;
            const float* clRow = (j > 0) ? cD + (((size_t)i * W + (j - 1)) * 32 + b) * G : 0;
            float*    cOut = cD + (((size_t)i * W + j) * 32 + b) * G;
            _Float16* hOut = hD + (((size_t)i * W + j) * 32 + b) * G;
            #pragma unroll
            for (int g0 = 0; g0 < G; g0 += 32) {
                const int g = g0 + lane;
                if (g < G) {
                    float pi  = pr[g];
                    float pf1 = pr[G + g];
                    float pf2 = pr[2 * G + g];
                    float pg  = pr[3 * G + g];
                    float po  = pr[4 * G + g];
                    float cu  = cuRow ? cuRow[g] : 0.0f;
                    float cl  = clRow ? clRow[g] : 0.0f;
                    float cn  = sigf(pf1) * cu + sigf(pf2) * cl + sigf(pi) * tanhf(pg);
                    cOut[g] = cn;
                    hOut[g] = (_Float16)(sigf(po) * tanhf(cn));
                }
            }
        }
        __syncthreads();
    }
}

// ---------------------------------------------------------------------------
// Average the 4 direction h-maps (unflipping by index) -> NCHW fp32 [32,G,H,W]
// ---------------------------------------------------------------------------
__global__ void avg_dirs(const _Float16* __restrict__ hbuf, float* __restrict__ out,
                         int H, int W, int G) {
    int total = 32 * G * H * W;
    int idx = blockIdx.x * blockDim.x + threadIdx.x;
    if (idx >= total) return;
    int w = idx % W;
    int h = (idx / W) % H;
    int g = (idx / (W * H)) % G;
    int b =  idx / (W * H * G);
    float s = 0.0f;
    for (int d = 0; d < 4; ++d) {
        int fH = (d >> 1) & 1, fW = d & 1;
        int li = fH ? (H - 1 - h) : h;
        int lj = fW ? (W - 1 - w) : w;
        s += (float)hbuf[(size_t)d * H * W * 32 * G +
                         (((size_t)li * W + lj) * 32 + b) * G + g];
    }
    out[idx] = 0.25f * s;
}

// ---------------------------------------------------------------------------
// Max over height: [32,80,9,31] -> [32,80,31]
// ---------------------------------------------------------------------------
__global__ void maxh_kernel(const float* __restrict__ in, float* __restrict__ out) {
    int total = 32 * 80 * 31;
    int idx = blockIdx.x * blockDim.x + threadIdx.x;
    if (idx >= total) return;
    int w = idx % 31;
    int c = (idx / 31) % 80;
    int b =  idx / (31 * 80);
    float m = -3.4e38f;
    for (int h = 0; h < 9; ++h)
        m = fmaxf(m, in[(((size_t)b * 80 + c) * 9 + h) * 31 + w]);
    out[idx] = m;
}

// ---------------------------------------------------------------------------
// Per-column FC: out[b,w,o] = sum_c mh[b,c,w] * fcw[o,c] + fcb[o]
// ---------------------------------------------------------------------------
__global__ void fc_kernel(const float* __restrict__ mh, const float* __restrict__ w80,
                          const float* __restrict__ bias, float* __restrict__ out) {
    int total = 32 * 31 * 80;
    int idx = blockIdx.x * blockDim.x + threadIdx.x;
    if (idx >= total) return;
    int o = idx % 80;
    int w = (idx / 80) % 31;
    int b =  idx / (80 * 31);
    float acc = bias[o];
    for (int c = 0; c < 80; ++c)
        acc += mh[((size_t)b * 80 + c) * 31 + w] * w80[o * 80 + c];
    out[idx] = acc;
}

// ---------------------------------------------------------------------------
struct LCfg { int Ci, Co, kh, kw, Hi, Wi, pool, H, W, G, Ktot, Kpad; };

extern "C" void kernel_launch(void* const* d_in, const int* in_sizes, int n_in,
                              void* d_out, int out_size, void* d_ws, size_t ws_size,
                              hipStream_t stream) {
    (void)in_sizes; (void)n_in; (void)out_size; (void)ws_size;
    static const LCfg LS[5] = {
        { 3,  8, 3, 3, 128, 286, 1, 63, 142, 16,  40,  64},
        {16, 24, 3, 3,  63, 142, 1, 30,  70, 32,  88,  96},
        {32, 40, 3, 3,  30,  70, 1, 14,  34, 48, 136, 160},
        {48, 56, 3, 3,  14,  34, 0, 12,  32, 64, 184, 192},
        {64, 72, 4, 2,  12,  32, 0,  9,  31, 80, 232, 256},
    };

    // ---- workspace carve-out (buffers reused across layers)
    char* ws = (char*)d_ws;
    size_t off = 0;
    auto carve = [&](size_t bytes) -> void* {
        void* p = ws + off;
        off = (off + bytes + 255) & ~(size_t)255;
        return p;
    };
    size_t maxAct = 0, maxHC = 0, maxNCHW = 0, wpackTot = 0;
    size_t wpOff[5];
    for (int l = 0; l < 5; ++l) {
        const LCfg& c = LS[l];
        size_t actE  = (size_t)c.H * c.W * 32 * c.Co;
        size_t hcE   = (size_t)4 * c.H * c.W * 32 * c.G;
        size_t nchwE = (size_t)32 * c.G * c.H * c.W;
        if (actE  > maxAct)  maxAct  = actE;
        if (hcE   > maxHC)   maxHC   = hcE;
        if (nchwE > maxNCHW) maxNCHW = nchwE;
        wpOff[l] = wpackTot;
        wpackTot += (size_t)4 * c.Kpad * 5 * c.G;   // == 4*KC*NT*512
    }
    _Float16* act   = (_Float16*)carve(maxAct * sizeof(_Float16));
    _Float16* hbuf  = (_Float16*)carve(maxHC * sizeof(_Float16));
    float*    cbuf  = (float*)   carve(maxHC * sizeof(float));
    float*    nchw  = (float*)   carve(maxNCHW * sizeof(float));
    _Float16* wpack = (_Float16*)carve(wpackTot * sizeof(_Float16));
    float*    mhTmp = (float*)   carve((size_t)32 * 80 * 31 * sizeof(float));

    // ---- pack MDLSTM weights (fp32 -> fp16, WMMA B-fragment layout)
    for (int l = 0; l < 5; ++l) {
        const LCfg& c = LS[l];
        int KC = c.Kpad / 32, Ntiles = 5 * c.G / 16;
        int total = 4 * KC * Ntiles * 512;
        pack_weights_frag<<<(total + BDIM - 1) / BDIM, BDIM, 0, stream>>>(
            (const float*)d_in[11 + 4 * l], (const float*)d_in[12 + 4 * l],
            (const float*)d_in[13 + 4 * l], wpack + wpOff[l],
            c.Co, c.G, c.Ktot, KC, Ntiles);
    }

    // ---- blocks 1..5
    for (int l = 0; l < 5; ++l) {
        const LCfg& c = LS[l];
        const float* convIn = (l == 0) ? (const float*)d_in[0] : nchw;
        int convTot = c.H * c.W * 32 * c.Co;
        conv_tanh_hwbc<<<(convTot + BDIM - 1) / BDIM, BDIM, 0, stream>>>(
            convIn, (const float*)d_in[1 + 2 * l], (const float*)d_in[2 + 2 * l],
            act, c.Ci, c.Co, c.kh, c.kw, c.Hi, c.Wi, c.H, c.W, c.pool);

        int nd = c.H + c.W - 1;
        for (int dd = 0; dd < nd; ++dd) {
            int ilo = dd - (c.W - 1); if (ilo < 0) ilo = 0;
            int ihi = dd < (c.H - 1) ? dd : (c.H - 1);
            int len = ihi - ilo + 1;
            dim3 grid(len, 4);
            const float* mdb = (const float*)d_in[14 + 4 * l];
            switch (l) {
                case 0: mdlstm_diag< 8, 16,  40,  64><<<grid, BDIM, 0, stream>>>(
                            act, wpack + wpOff[l], mdb, hbuf, cbuf, c.H, c.W, dd, ilo); break;
                case 1: mdlstm_diag<24, 32,  88,  96><<<grid, BDIM, 0, stream>>>(
                            act, wpack + wpOff[l], mdb, hbuf, cbuf, c.H, c.W, dd, ilo); break;
                case 2: mdlstm_diag<40, 48, 136, 160><<<grid, BDIM, 0, stream>>>(
                            act, wpack + wpOff[l], mdb, hbuf, cbuf, c.H, c.W, dd, ilo); break;
                case 3: mdlstm_diag<56, 64, 184, 192><<<grid, BDIM, 0, stream>>>(
                            act, wpack + wpOff[l], mdb, hbuf, cbuf, c.H, c.W, dd, ilo); break;
                case 4: mdlstm_diag<72, 80, 232, 256><<<grid, BDIM, 0, stream>>>(
                            act, wpack + wpOff[l], mdb, hbuf, cbuf, c.H, c.W, dd, ilo); break;
            }
        }

        int avgTot = 32 * c.G * c.H * c.W;
        avg_dirs<<<(avgTot + BDIM - 1) / BDIM, BDIM, 0, stream>>>(
            hbuf, nchw, c.H, c.W, c.G);
    }

    // ---- collapse: max over H, then per-column FC
    maxh_kernel<<<(32 * 80 * 31 + BDIM - 1) / BDIM, BDIM, 0, stream>>>(nchw, mhTmp);
    fc_kernel<<<(32 * 31 * 80 + BDIM - 1) / BDIM, BDIM, 0, stream>>>(
        mhTmp, (const float*)d_in[31], (const float*)d_in[32], (float*)d_out);
}
